// CommutatorConv2d_31954556682526
// MI455X (gfx1250) — matrix operand order repaired and checked by reading, not compiled, not run on hardware
//
#include <hip/hip_runtime.h>

typedef __attribute__((ext_vector_type(16))) _Float16     v16h;
typedef __attribute__((ext_vector_type(8)))  _Float16     v8h;
typedef __attribute__((ext_vector_type(8)))  float        v8f;
typedef __attribute__((ext_vector_type(4)))  unsigned int u32x4;
typedef __attribute__((ext_vector_type(8)))  int          i32x8;
typedef __attribute__((ext_vector_type(4)))  int          i32x4;

#define CI 128
#define CO 256
#define HH 56
#define WW 56
#define BATCH 32
#define HWP (HH*WW)        // 3136
#define KKTOT (CI*9)       // 1152
#define NTOT (BATCH*HWP)   // 100352

#define MT 128
#define NT 128
#define KT 32
#define ROWA 40            // padded LDS row (halves): 80B; TDM pad 16dw+4dw reproduces this
#define ROWB 40

#define LAMBDA_C 0.0f
#define LAMBDA_A 1.0f

#if __has_builtin(__builtin_amdgcn_tensor_load_to_lds)
#define USE_TDM 1
#else
#define USE_TDM 0
#endif

// ---------------------------------------------------------------------------
// Prep: w_eff[o, (r*3+s)*CI + ci] = (la+lc)*rowsum(w)[o,ci,s] + (la-lc)*colsum(w)[o,ci,r]
// ---------------------------------------------------------------------------
__global__ void weff_prep_kernel(const float* __restrict__ w, _Float16* __restrict__ weff) {
    int idx = blockIdx.x * 256 + threadIdx.x;   // o*CI + ci
    if (idx >= CO * CI) return;
    int o  = idx >> 7;
    int ci = idx & 127;
    const float* wp = w + (size_t)idx * 9;
    float v[9];
#pragma unroll
    for (int i = 0; i < 9; ++i) v[i] = wp[i];
    float krow[3], kcol[3];
#pragma unroll
    for (int s = 0; s < 3; ++s) krow[s] = v[s] + v[3 + s] + v[6 + s];
#pragma unroll
    for (int r = 0; r < 3; ++r) kcol[r] = v[3 * r] + v[3 * r + 1] + v[3 * r + 2];
    _Float16* op = weff + (size_t)o * KKTOT + ci;
#pragma unroll
    for (int r = 0; r < 3; ++r)
#pragma unroll
        for (int s = 0; s < 3; ++s)
            op[(r * 3 + s) * CI] = (_Float16)((LAMBDA_A + LAMBDA_C) * krow[s] +
                                              (LAMBDA_A - LAMBDA_C) * kcol[r]);
}

// ---------------------------------------------------------------------------
// Implicit-GEMM conv, double-buffered. A tile via TDM (tensor_load_to_lds),
// B tile register-staged gather with f32->f16 convert.
// ---------------------------------------------------------------------------
__global__ __launch_bounds__(256) void commconv_wmma_kernel(
        const float*    __restrict__ x,
        const _Float16* __restrict__ weff,
        const float*    __restrict__ bias,
        float*          __restrict__ out) {
    __shared__ _Float16 At[2][MT * ROWA];   // 2 x 10240 B, natural K order, padded rows
    __shared__ _Float16 Bt[2][NT * ROWB];   // 2 x 10240 B

    const int tid  = threadIdx.x;
    const int lane = tid & 31;
    const int wv   = tid >> 5;
    const int m0   = blockIdx.x * MT;       // gridDim.x = 2
    const int n0   = blockIdx.y * NT;       // gridDim.y = 784

    // ---- B loader: thread -> (pixel np, k-half) ----
    const int np    = tid & 127;
    const int khalf = tid >> 7;
    const int p     = n0 + np;
    const int pb    = p / HWP;
    const int phw   = p - pb * HWP;
    const int ph    = phw / WW;
    const int pw    = phw - ph * WW;
    const float* xrow = x + (long)pb * CI * HWP + (long)khalf * 16 * HWP
                          + (long)(ph - 1) * WW + (pw - 1);

    // ---- wave fragment mapping ----
    const int m_off  = (wv & 3) * 32;       // two A frags: m_off, m_off+16
    const int n_base = (wv >> 2) * 64;      // four B frags
    const int frow   = lane & 15;
    const int lhalf  = lane >> 4;

    v8f acc[2][4];
#pragma unroll
    for (int i = 0; i < 2; ++i)
#pragma unroll
        for (int f = 0; f < 4; ++f) acc[i][f] = v8f{};

    uint4 sb0, sb1;                          // register-staged B fragment

    auto load_stage_b = [&](int kk) {
        int rs = kk >> 7;                    // uniform: SALU
        int cb = kk & 127;
        int r  = (rs * 11) >> 5;             // rs/3 for rs<9
        int sc = rs - 3 * r;
        int ih = ph + r - 1;
        int iw = pw + sc - 1;
        const float* q = xrow + (long)cb * HWP + r * WW + sc;
        union { _Float16 h[16]; uint4 v[2]; } pk;
        if ((unsigned)ih < (unsigned)HH && (unsigned)iw < (unsigned)WW) {
#pragma unroll
            for (int j = 0; j < 16; ++j)     // immediate offsets j*HWP*4
                pk.h[j] = (_Float16)q[(long)j * HWP];
        } else {
#pragma unroll
            for (int j = 0; j < 16; ++j) pk.h[j] = (_Float16)0.0f;
        }
        sb0 = pk.v[0];
        sb1 = pk.v[1];
    };

    auto store_stage_b = [&](int buf) {
        uint4* bd = (uint4*)(&Bt[buf][np * ROWB + khalf * 16]);
        bd[0] = sb0;
        bd[1] = sb1;
    };

#if USE_TDM
    // ---- A tile via Tensor Data Mover: 128x32 halves, stride 1152, pad 16dw+4dw ----
    auto issue_a = [&](int buf, int kk) {
        if (wv == 0) {
            unsigned long long ga =
                (unsigned long long)(const void*)(weff + (size_t)m0 * KKTOT + kk);
            unsigned int lds0 = (unsigned int)(unsigned long long)&At[buf][0];
            u32x4 g0;
            g0[0] = 1u;                                   // count=1, no gather
            g0[1] = lds0;                                 // lds_addr
            g0[2] = (unsigned int)ga;                     // global_addr[31:0]
            g0[3] = (unsigned int)(ga >> 32) | (2u << 30);// addr[56:32] | type=2
            i32x8 g1;
            g1[0] = (1 << 16)      // data_size = 2 bytes
                  | (1 << 20)      // pad_enable
                  | (3 << 22)      // pad_interval: 16 DWORDs (= one 32-half row)
                  | (3 << 25);     // pad_amount:   4 DWORDs (= 8 halves)
            g1[1] = (int)((unsigned)KKTOT << 16);         // tensor_dim0 = 1152
            g1[2] = (int)((unsigned)CO << 16);            // tensor_dim1 = 256
            g1[3] = (int)((unsigned)KT << 16);            // tile_dim0 = 32
            g1[4] = MT;                                   // tile_dim1 = 128
            g1[5] = KKTOT;                                // tensor_dim0_stride = 1152
            g1[6] = 0;
            g1[7] = 0;
            i32x4 zz4 = {0, 0, 0, 0};
            i32x8 zz8 = {0, 0, 0, 0, 0, 0, 0, 0};
            __builtin_amdgcn_tensor_load_to_lds(g0, g1, zz4, zz4, zz8, 0);
        }
    };
#else
    // ---- fallback: register-staged A (2 threads per row, 32B each) ----
    const int arow  = tid >> 1;
    const int acolh = tid & 1;
    const _Float16* weffp = weff + (size_t)(m0 + arow) * KKTOT + acolh * 16;
    uint4 sa0, sa1;
    auto load_stage_a = [&](int kk) {
        const uint4* s4 = (const uint4*)(weffp + kk);
        sa0 = s4[0];
        sa1 = s4[1];
    };
    auto store_stage_a = [&](int buf) {
        uint4* ad = (uint4*)(&At[buf][arow * ROWA + acolh * 16]);
        ad[0] = sa0;
        ad[1] = sa1;
    };
#endif

    // per-lane A fragment: halves K(lhalf*16..+15) contiguous
    auto make_a = [&](int buf, int mrow) -> v16h {
        const _Float16* rowp = &At[buf][mrow * ROWA];
        v8h lo = *(const v8h*)(rowp + lhalf * 16);
        v8h hi = *(const v8h*)(rowp + lhalf * 16 + 8);
        v16h a;
#pragma unroll
        for (int e = 0; e < 8; ++e) { a[e] = lo[e]; a[e + 8] = hi[e]; }
        return a;
    };

    // ---- prologue ----
#if USE_TDM
    issue_a(0, 0);
#else
    load_stage_a(0);
    store_stage_a(0);
#endif
    load_stage_b(0);
    store_stage_b(0);
#if USE_TDM
    __builtin_amdgcn_s_wait_tensorcnt(0);
#endif
    __syncthreads();

    int buf = 0;
    for (int kk = 0; kk < KKTOT; kk += KT, buf ^= 1) {
        const bool more = (kk + KT) < KKTOT;
        if (more) {
#if USE_TDM
            issue_a(buf ^ 1, kk + KT);      // DMA overlaps the WMMAs below
#else
            load_stage_a(kk + KT);
#endif
            load_stage_b(kk + KT);
        }

        v16h a0 = make_a(buf, m_off + frow);
        v16h a1 = make_a(buf, m_off + 16 + frow);
#pragma unroll
        for (int f = 0; f < 4; ++f) {
            v16h bf = *(const v16h*)(&Bt[buf][(n_base + f * 16 + frow) * ROWB + lhalf * 16]);
            acc[0][f] = __builtin_amdgcn_wmma_f32_16x16x32_f16(
                false, a0, false, bf, (short)0, acc[0][f], false, false);
            acc[1][f] = __builtin_amdgcn_wmma_f32_16x16x32_f16(
                false, a1, false, bf, (short)0, acc[1][f], false, false);
        }

        if (more) {
#if !USE_TDM
            store_stage_a(buf ^ 1);
#endif
            store_stage_b(buf ^ 1);
        }
#if USE_TDM
        __builtin_amdgcn_s_wait_tensorcnt(0);   // no-op for non-issuing waves
#endif
        __syncthreads();
    }

    // ---- epilogue: D layout N = lane%16, M = vgpr + lhalf*8 ----
#pragma unroll
    for (int i = 0; i < 2; ++i) {
        const int och = m0 + m_off + i * 16 + lhalf * 8;
#pragma unroll
        for (int f = 0; f < 4; ++f) {
            int p2  = n0 + n_base + f * 16 + frow;
            int b2  = p2 / HWP;
            int hw2 = p2 - b2 * HWP;
            long ob = (long)b2 * CO * HWP + hw2;
#pragma unroll
            for (int j = 0; j < 8; ++j)
                out[ob + (long)(och + j) * HWP] = acc[i][f][j] + bias[och + j];
        }
    }
}

// ---------------------------------------------------------------------------
extern "C" void kernel_launch(void* const* d_in, const int* in_sizes, int n_in,
                              void* d_out, int out_size, void* d_ws, size_t ws_size,
                              hipStream_t stream) {
    (void)in_sizes; (void)n_in; (void)out_size; (void)ws_size;
    const float* x    = (const float*)d_in[0];
    const float* w    = (const float*)d_in[1];
    const float* bias = (const float*)d_in[2];
    float*       out  = (float*)d_out;
    _Float16*    weff = (_Float16*)d_ws;   // needs 589,824 bytes

    weff_prep_kernel<<<dim3((CO * CI + 255) / 256), dim3(256), 0, stream>>>(w, weff);

    dim3 grid(CO / MT, NTOT / NT);         // (2, 784)
    commconv_wmma_kernel<<<grid, dim3(256), 0, stream>>>(x, weff, bias, out);
}
